// MorphModel_59554016526401
// MI455X (gfx1250) — matrix operand order repaired and checked by reading, not compiled, not run on hardware
//
#include <hip/hip_runtime.h>
#include <hip/hip_bf16.h>

#define E_   16
#define B_   32768
#define DIN  256
#define DH   512
#define DOUT 256
#define MTILE 64          // tokens per expert-block
#define TPE   512         // max tiles per expert = B_/MTILE

#define XSTR 264          // Xa row stride (shorts): 132 dwords -> conflict-free b128
#define HSTR 520          // Ha row stride (shorts): 260 dwords
#define WSTR 40           // W-slab column stride (shorts): 20 dwords

typedef __attribute__((ext_vector_type(16))) __bf16 v16bf;
typedef __attribute__((ext_vector_type(16))) short  v16s;
typedef __attribute__((ext_vector_type(8)))  short  v8s;
typedef __attribute__((ext_vector_type(8)))  float  v8f;

__device__ __forceinline__ unsigned short f32_bf16(float f) {
    unsigned int u = __float_as_uint(f);
    u += 0x7FFFu + ((u >> 16) & 1u);      // round-to-nearest-even
    return (unsigned short)(u >> 16);
}

__device__ __forceinline__ v16bf cat16(v8s lo, v8s hi) {
    v16s c = __builtin_shufflevector(lo, hi, 0,1,2,3,4,5,6,7,8,9,10,11,12,13,14,15);
    return __builtin_bit_cast(v16bf, c);
}

// CDNA5 async DMA: global -> LDS, 16B per lane, tracked by ASYNCcnt.
template <int NC>
__device__ __forceinline__ void issue_slab(const unsigned short* __restrict__ slab,
                                           unsigned short* wsbuf, int tid) {
#pragma unroll
    for (int c4 = 0; c4 < NC; ++c4) {
        int c = tid + c4 * 512;
        int n = c >> 2, k16 = c & 3;
        unsigned int lds = (unsigned int)(uintptr_t)&wsbuf[n * WSTR + k16 * 8];
        const unsigned short* g = slab + c * 8;   // 16B chunk c
        asm volatile("global_load_async_to_lds_b128 %0, %1, off"
                     :: "v"(lds), "v"(g) : "memory");
    }
}
__device__ __forceinline__ void wait_async0() {
    asm volatile("s_wait_asynccnt 0x0" ::: "memory");
}

// ---------------------------------------------------------------- zero
__global__ void zero_kernel(float* out, int n, int* counts) {
    int i = blockIdx.x * blockDim.x + threadIdx.x;
    if (i < E_) counts[i] = 0;
    for (int j = i; j < n; j += gridDim.x * blockDim.x) out[j] = 0.f;
}

// ---------------------------------------------------------------- weight prep
// W1t: [e][kt<8][n<512][k<32] bf16 ; W2t: [e][kt<16][n<256][k<32] bf16
__global__ void prep_w_kernel(const float* __restrict__ W1, const float* __restrict__ W2,
                              unsigned short* __restrict__ W1t,
                              unsigned short* __restrict__ W2t) {
    int stride = gridDim.x * blockDim.x;
    for (int idx = blockIdx.x * blockDim.x + threadIdx.x; idx < (E_ * DIN * DH);
         idx += stride) {
        {   // W1t
            int k  = idx & 31;
            int n  = (idx >> 5) & (DH - 1);
            int kt = (idx >> 14) & 7;
            int e  = idx >> 17;
            W1t[idx] = f32_bf16(W1[((e * 8 + kt) * 32 + k) * DH + n]);
        }
        {   // W2t
            int k  = idx & 31;
            int n  = (idx >> 5) & (DOUT - 1);
            int kt = (idx >> 13) & 15;
            int e  = idx >> 17;
            W2t[idx] = f32_bf16(W2[((e * 16 + kt) * 32 + k) * DOUT + n]);
        }
    }
}

// ---------------------------------------------------------------- gating
__global__ __launch_bounds__(256)
void gate_kernel(const float* __restrict__ x, const float* __restrict__ Wg,
                 const float* __restrict__ bg,
                 int* counts, int* toklist, float* wtlist) {
    __shared__ float WgS[DIN * E_];
    for (int i = threadIdx.x; i < DIN * E_; i += 256) WgS[i] = Wg[i];
    __syncthreads();

    int wave = threadIdx.x >> 5, lane = threadIdx.x & 31;
    int t = blockIdx.x * 8 + wave;

    float acc[E_];
#pragma unroll
    for (int e = 0; e < E_; ++e) acc[e] = 0.f;
#pragma unroll
    for (int j = 0; j < DIN / 32; ++j) {
        int i = j * 32 + lane;
        float xv = x[t * DIN + i];
        const float* wr = &WgS[i * E_];
#pragma unroll
        for (int e = 0; e < E_; ++e) acc[e] += xv * wr[e];
    }
#pragma unroll
    for (int s = 16; s >= 1; s >>= 1)
#pragma unroll
        for (int e = 0; e < E_; ++e) acc[e] += __shfl_xor(acc[e], s, 32);

    if (lane == 0) {
        float l0 = -1e30f, l1 = -1e30f; int e0 = 0, e1 = 1;
#pragma unroll
        for (int e = 0; e < E_; ++e) {
            float v = acc[e] + bg[e];
            if (v > l0)      { l1 = l0; e1 = e0; l0 = v; e0 = e; }
            else if (v > l1) { l1 = v;  e1 = e; }
        }
        float w0 = 1.f / (1.f + __expf(l1 - l0));   // renormalized top-2 softmax
        float w1 = 1.f - w0;
        int p0 = atomicAdd(&counts[e0], 1);
        toklist[e0 * B_ + p0] = t; wtlist[e0 * B_ + p0] = w0;
        int p1 = atomicAdd(&counts[e1], 1);
        toklist[e1 * B_ + p1] = t; wtlist[e1 * B_ + p1] = w1;
    }
}

// ---------------------------------------------------------------- expert MLP
__global__ __launch_bounds__(512)
void expert_kernel(const float* __restrict__ x,
                   const unsigned short* __restrict__ W1t,
                   const float* __restrict__ b1,
                   const unsigned short* __restrict__ W2t,
                   const float* __restrict__ b2,
                   const int* __restrict__ counts,
                   const int* __restrict__ toklist,
                   const float* __restrict__ wtlist,
                   float* __restrict__ out) {
    int e    = blockIdx.x >> 9;          // /TPE
    int tile = blockIdx.x & (TPE - 1);
    int ne = counts[e];
    if (tile * MTILE >= ne) return;      // block-uniform exit (EXEC stays full)

    extern __shared__ char smem[];
    unsigned short* Xa  = (unsigned short*)(smem);             // 64 x XSTR  (33792 B)
    unsigned short* Ha  = (unsigned short*)(smem + 33792);     // 64 x HSTR  (66560 B)
    unsigned short* Ws0 = (unsigned short*)(smem + 100352);    // 512 x WSTR (40960 B)
    unsigned short* Ws1 = (unsigned short*)(smem + 141312);    // 512 x WSTR (40960 B)
    int*   tokS = (int*)  (smem + 182272);                     // 64 ints
    float* wtS  = (float*)(smem + 182528);                     // 64 floats

    const int tid  = threadIdx.x;
    const int wave = tid >> 5, lane = tid & 31;
    const int grp  = lane >> 4, ln16 = lane & 15;

    const unsigned short* W1base = W1t + (size_t)e * 8  * DH   * 32;
    const unsigned short* W2base = W2t + (size_t)e * 16 * DOUT * 32;

    // kick off async DMA of the first GEMM1 weight slab immediately
    issue_slab<4>(W1base, Ws0, tid);

    if (tid < MTILE) {
        int idx = tile * MTILE + tid;
        bool v = idx < ne;
        int src = e * B_ + (v ? idx : tile * MTILE);
        tokS[tid] = toklist[src];
        wtS[tid]  = v ? wtlist[src] : 0.f;
    }
    __syncthreads();

    // stage X tile 64x256 (overlaps with slab-0 async DMA)
    for (int i = tid; i < MTILE * (DIN / 2); i += 512) {
        int r = i >> 7, c2 = i & 127;
        float2 xv = *(const float2*)&x[tokS[r] * DIN + c2 * 2];
        unsigned int p = (unsigned int)f32_bf16(xv.x) |
                         ((unsigned int)f32_bf16(xv.y) << 16);
        *(unsigned int*)&Xa[r * XSTR + c2 * 2] = p;
    }

    // wave -> 2 M-tiles x 4 N-tiles (GEMM1): B frags reused across both M-tiles
    const int mh = wave & 1;             // rows [mh*32, mh*32+32)
    const int nq = wave >> 1;            // cols [nq*64, nq*64+64)
    const int mrow0 = mh * 32 + ln16;    // M-tile 0 row
    // (M-tile 1 rows = mrow0 + 16)

    v8f acc1[2][4];
#pragma unroll
    for (int mi = 0; mi < 2; ++mi)
#pragma unroll
        for (int j = 0; j < 4; ++j) acc1[mi][j] = (v8f){};

#pragma unroll
    for (int kt = 0; kt < DIN / 32; ++kt) {
        wait_async0();                   // slab kt landed
        __syncthreads();                 // all waves: slab ready, prev buffer free
        if (kt + 1 < DIN / 32)
            issue_slab<4>(W1base + (kt + 1) * DH * 32, (kt & 1) ? Ws0 : Ws1, tid);
        else
            issue_slab<2>(W2base, Ws0, tid);   // early GEMM2 slab 0
        const unsigned short* cur = (kt & 1) ? Ws1 : Ws0;

        v16bf a[2];
#pragma unroll
        for (int mi = 0; mi < 2; ++mi) {
            const unsigned short* ap = &Xa[(mrow0 + mi * 16) * XSTR + kt * 32 + grp * 8];
            a[mi] = cat16(*(const v8s*)ap, *(const v8s*)(ap + 16));
        }
        v16bf bf[4];
#pragma unroll
        for (int j = 0; j < 4; ++j) {
            const unsigned short* bp = &cur[(nq * 64 + j * 16 + ln16) * WSTR + grp * 8];
            bf[j] = cat16(*(const v8s*)bp, *(const v8s*)(bp + 16));
        }
#pragma unroll
        for (int j = 0; j < 4; ++j)
#pragma unroll
            for (int mi = 0; mi < 2; ++mi)
                acc1[mi][j] = __builtin_amdgcn_wmma_f32_16x16x32_bf16(
                    false, a[mi], false, bf[j], (short)0, acc1[mi][j], false, false);
    }

    // bias + ReLU -> Ha (bf16); C layout: VGPR r -> M = r + 8*grp, N = ln16
#pragma unroll
    for (int j = 0; j < 4; ++j) {
        int n = nq * 64 + j * 16 + ln16;
        float bias = b1[e * DH + n];
#pragma unroll
        for (int mi = 0; mi < 2; ++mi)
#pragma unroll
            for (int r = 0; r < 8; ++r) {
                int m = mh * 32 + mi * 16 + grp * 8 + r;
                float v = acc1[mi][j][r] + bias;
                Ha[m * HSTR + n] = f32_bf16(v > 0.f ? v : 0.f);
            }
    }

    // ---------------- GEMM2: Ha[64x512] @ W2[e][512x256] -> out ----------------
    // wave -> 2 M-tiles x 2 N-tiles
    const int nq2 = wave >> 1;           // cols [nq2*32, nq2*32+32)
    v8f acc2[2][2];
#pragma unroll
    for (int mi = 0; mi < 2; ++mi)
#pragma unroll
        for (int j = 0; j < 2; ++j) acc2[mi][j] = (v8f){};

#pragma unroll
    for (int kt = 0; kt < DH / 32; ++kt) {
        wait_async0();
        __syncthreads();                 // also orders Ha writes vs reads (kt==0)
        if (kt + 1 < DH / 32)
            issue_slab<2>(W2base + (kt + 1) * DOUT * 32, (kt & 1) ? Ws0 : Ws1, tid);
        const unsigned short* cur = (kt & 1) ? Ws1 : Ws0;

        v16bf a[2];
#pragma unroll
        for (int mi = 0; mi < 2; ++mi) {
            const unsigned short* ap = &Ha[(mrow0 + mi * 16) * HSTR + kt * 32 + grp * 8];
            a[mi] = cat16(*(const v8s*)ap, *(const v8s*)(ap + 16));
        }
        v16bf bf[2];
#pragma unroll
        for (int j = 0; j < 2; ++j) {
            const unsigned short* bp = &cur[(nq2 * 32 + j * 16 + ln16) * WSTR + grp * 8];
            bf[j] = cat16(*(const v8s*)bp, *(const v8s*)(bp + 16));
        }
#pragma unroll
        for (int j = 0; j < 2; ++j)
#pragma unroll
            for (int mi = 0; mi < 2; ++mi)
                acc2[mi][j] = __builtin_amdgcn_wmma_f32_16x16x32_bf16(
                    false, a[mi], false, bf[j], (short)0, acc2[mi][j], false, false);
    }

    // bias + routing weight, scatter-add
#pragma unroll
    for (int j = 0; j < 2; ++j) {
        int n = nq2 * 32 + j * 16 + ln16;
        float bias = b2[e * DOUT + n];
#pragma unroll
        for (int mi = 0; mi < 2; ++mi)
#pragma unroll
            for (int r = 0; r < 8; ++r) {
                int m = mh * 32 + mi * 16 + grp * 8 + r;
                float v = acc2[mi][j][r] + bias;
                atomicAdd(&out[tokS[m] * DOUT + n], wtS[m] * v);
            }
    }
}

// ---------------------------------------------------------------- launcher
extern "C" void kernel_launch(void* const* d_in, const int* in_sizes, int n_in,
                              void* d_out, int out_size, void* d_ws, size_t ws_size,
                              hipStream_t stream) {
    const float* x  = (const float*)d_in[0];
    const float* Wg = (const float*)d_in[1];
    const float* bg = (const float*)d_in[2];
    const float* W1 = (const float*)d_in[3];
    const float* b1 = (const float*)d_in[4];
    const float* W2 = (const float*)d_in[5];
    const float* b2 = (const float*)d_in[6];
    float* out = (float*)d_out;

    char* ws = (char*)d_ws;
    int*   counts  = (int*)ws;                                   // 256 B region
    int*   toklist = (int*)(ws + 256);                           // 2 MB
    float* wtlist  = (float*)(ws + 256 + (size_t)E_ * B_ * 4);   // 2 MB
    unsigned short* W1t = (unsigned short*)(ws + 256 + (size_t)E_ * B_ * 8);          // 4 MB
    unsigned short* W2t = (unsigned short*)(ws + 256 + (size_t)E_ * B_ * 8 + 4194304);// 4 MB

    zero_kernel<<<4096, 256, 0, stream>>>(out, B_ * DOUT, counts);
    prep_w_kernel<<<4096, 256, 0, stream>>>(W1, W2, W1t, W2t);
    gate_kernel<<<B_ / 8, 256, 0, stream>>>(x, Wg, bg, counts, toklist, wtlist);

    size_t lds = 182784;  // Xa+Ha+Ws0+Ws1+tok/wt (one 512-thread group per WGP)
    expert_kernel<<<E_ * TPE, 512, lds, stream>>>(x, W1t, b1, W2t, b2,
                                                  counts, toklist, wtlist, out);
}
